// RefinedPyrUNetLoss_15857019257544
// MI455X (gfx1250) — matrix operand order repaired and checked by reading, not compiled, not run on hardware
//
#include <hip/hip_runtime.h>
#include <hip/hip_bf16.h>
#include <math.h>

typedef __attribute__((ext_vector_type(16))) _Float16 v16h;
typedef __attribute__((ext_vector_type(8)))  float    v8f;

namespace {
constexpr int   kHW    = 256 * 256;   // H*W
constexpr int   kChT   = 12 * kHW;    // channel stride (D*H*W) for target & pred
constexpr int   kBstT  = 4  * kChT;   // target batch stride
constexpr int   kBstP  = 32 * kChT;   // pred batch stride
constexpr int   kNOut  = 1805;        // 5*19*19
constexpr int   kNTiles = 113;        // ceil(1805/16)
constexpr float kEps   = 1e-6f;
}

__global__ void __launch_bounds__(32) k_zero(float* ws) {
  if (threadIdx.x < 2) ws[threadIdx.x] = 0.f;
}

__global__ void __launch_bounds__(32) k_final(const float* ws, float* out) {
  if (threadIdx.x == 0) out[0] = -2.0f * ws[0] / fmaxf(ws[1], kEps);
}

// One block = one M-tile of 16 patches. 128 threads = 4 waves (wave32).
// GEMM1: emb(16x32) @ W1(32x256) -> h in LDS (f16). One WMMA per 16x16 tile (K=32).
// GEMM2: h(16x256) @ W2(256x1805), 8 WMMA per N-tile; epilogue fuses bias,
// sigmoid, max-pooled target/ignore masks and the dice accumulators.
__global__ void __launch_bounds__(128, 2) k_fused(
    const float* __restrict__ target, const float* __restrict__ pred,
    const float* __restrict__ W1v, const float* __restrict__ b1v,
    const float* __restrict__ W2v, const float* __restrict__ b2v,
    float* __restrict__ ws) {
  __shared__ _Float16 hLds[16][256];     // 8 KB staged activations
  __shared__ float centerLds[16];
  __shared__ float validLds[16];
  __shared__ int   gtBaseLds[16];
  __shared__ int   predBaseLds[16];
  __shared__ float blkNum, blkDen;

  const int tid  = threadIdx.x;
  const int lane = tid & 31;
  const int wave = tid >> 5;
  const int hi16 = lane >> 4;            // 0 | 1 (half-wave)
  const int l15  = lane & 15;

  if (tid == 0) { blkNum = 0.f; blkDen = 0.f; }

  // ---- per-patch staging: centers, validity, patch origins ----
  if (tid < 16) {
    const int pidx = blockIdx.x * 16 + tid;          // 0..9999
    const int b  = pidx / 5000;
    const int r0 = pidx - b * 5000;
    const int pd = r0 / 625;
    const int r1 = r0 - pd * 625;
    const int ph = r1 / 25;
    const int pw = r1 - ph * 25;
    const int y0 = 2 + ph * 9, x0 = 2 + pw * 9;      // patch origin (gt coords)
    const int gbase = b * kBstT + pd * kHW + y0 * 256 + x0;
    const int cidx  = gbase + 2 * kHW + 19 * 256 + 19;   // center = origin + REAL//2
    const float c  = target[cidx];
    const float a1 = target[cidx + kChT];
    const float a2 = target[cidx + 2 * kChT];
    const float a3 = target[cidx + 3 * kChT];
    const float onb = fmaxf(fmaxf(1.f - a1, 1.f - a2), 1.f - a3);
    centerLds[tid] = c;
    validLds[tid]  = ((c != 0.f) && (onb != 1.f)) ? 1.f : 0.f;
    gtBaseLds[tid] = gbase;
    predBaseLds[tid] = b * kBstP + (2 + pd) * kHW + (21 + ph * 9) * 256 + (21 + pw * 9);
  }
  __syncthreads();

  // ---- GEMM1: h = relu(emb @ W1 + b1) ----
  // A fragment (16x32 f16): lane l, half h -> row l&15, K=(h&7)+((h>>3)<<4)+((l>>4)<<3)
  v16h aEmb;
  {
    const int pbase = predBaseLds[l15];
#pragma unroll
    for (int h = 0; h < 16; ++h) {
      const int k = (h & 7) + ((h >> 3) << 4) + (hi16 << 3);
      aEmb[h] = (_Float16)pred[pbase + k * kChT];
    }
  }
#pragma unroll
  for (int i = 0; i < 4; ++i) {
    const int t  = wave * 4 + i;       // hidden tile 0..15
    const int nn = t * 16 + l15;       // hidden unit
    v16h bW1;                          // B fragment: col l&15, K = (h&7)+((h>>3)<<4)+((l>>4)<<3)
#pragma unroll
    for (int h = 0; h < 16; ++h)
      bW1[h] = (_Float16)W1v[((h & 7) + ((h >> 3) << 4) + (hi16 << 3)) * 256 + nn];
    v8f acc = {};
    acc = __builtin_amdgcn_wmma_f32_16x16x32_f16(false, aEmb, false, bW1,
                                                 (short)0, acc, false, false);
    const float bias = b1v[nn];
#pragma unroll
    for (int r = 0; r < 8; ++r) {
      const int mm = r + (hi16 << 3);  // C/D layout: row = r + ((l>>4)<<3)
      hLds[mm][nn] = (_Float16)fmaxf(acc[r] + bias, 0.f);
    }
  }
  __syncthreads();

  // ---- A fragments of h for GEMM2, built once, reused for all N-tiles ----
  v16h aH[8];
#pragma unroll
  for (int kk = 0; kk < 8; ++kk) {
#pragma unroll
    for (int h = 0; h < 16; ++h) {
      const int k = kk * 32 + (h & 7) + ((h >> 3) << 4) + (hi16 << 3);
      aH[kk][h] = hLds[l15][k];
    }
  }

  // ---- GEMM2 + fused epilogue ----
  float numAcc = 0.f, denAcc = 0.f;
  for (int j = wave; j < kNTiles; j += 4) {
    const int n   = j * 16 + l15;            // output element (padded to 1808)
    const bool inR = (n < kNOut);
    v8f acc = {};
#pragma unroll
    for (int kk = 0; kk < 8; ++kk) {
      const int kb = kk * 32 + (hi16 << 3);
      v16h bW2;
#pragma unroll
      for (int h = 0; h < 16; ++h)
        bW2[h] = inR ? (_Float16)W2v[(kb + (h & 7) + ((h >> 3) << 4)) * kNOut + n] : (_Float16)0.f;
      acc = __builtin_amdgcn_wmma_f32_16x16x32_f16(false, aH[kk], false, bW2,
                                                   (short)0, acc, false, false);
    }
    if (inR) {
      const float bias = b2v[n];
      const int oz = n / 361;
      const int rz = n - oz * 361;
      const int oy = rz / 19;
      const int ox = rz - oy * 19;
      const int poff = oz * kHW + (oy * 2) * 256 + (ox * 2);  // 1x2x2 pool window
#pragma unroll
      for (int r = 0; r < 8; ++r) {
        const int mm = r + (hi16 << 3);                 // patch row in tile
        const float p  = 1.f / (1.f + __expf(-(acc[r] + bias)));
        const float c  = centerLds[mm];
        const float vm = validLds[mm];
        const int gb = gtBaseLds[mm] + poff;
        const float g00 = target[gb];
        const float g01 = target[gb + 1];
        const float g10 = target[gb + 256];
        const float g11 = target[gb + 257];
        const bool me  = (g00 != c) | (g01 != c) | (g10 != c) | (g11 != c);
        const bool ign = (g00 == 0.f) | (g01 == 0.f) | (g10 == 0.f) | (g11 == 0.f);
        const float msk = ign ? 0.f : vm;
        const float pm = p * msk;
        const float tm = me ? msk : 0.f;
        numAcc = fmaf(pm, tm, numAcc);
        denAcc = fmaf(pm, pm, fmaf(tm, tm, denAcc));
      }
    }
  }

  atomicAdd(&blkNum, numAcc);
  atomicAdd(&blkDen, denAcc);
  __syncthreads();
  if (tid == 0) {
    atomicAdd(&ws[0], blkNum);
    atomicAdd(&ws[1], blkDen);
  }
}

extern "C" void kernel_launch(void* const* d_in, const int* in_sizes, int n_in,
                              void* d_out, int out_size, void* d_ws, size_t ws_size,
                              hipStream_t stream) {
  const float* target = (const float*)d_in[0];
  const float* pred   = (const float*)d_in[1];
  const float* W1     = (const float*)d_in[2];
  const float* b1     = (const float*)d_in[3];
  const float* W2     = (const float*)d_in[4];
  const float* b2     = (const float*)d_in[5];
  float* out = (float*)d_out;
  float* ws  = (float*)d_ws;

  k_zero <<<1, 32, 0, stream>>>(ws);
  k_fused<<<625, 128, 0, stream>>>(target, pred, W1, b1, W2, b2, ws);
  k_final<<<1, 32, 0, stream>>>(ws, out);
}
